// WeightedConformers_67130338836859
// MI455X (gfx1250) — compile-verified, matrix operand order checked by reading, not run on hardware
//
#include <hip/hip_runtime.h>
#include <hip/hip_bf16.h>

// ---------------------------------------------------------------------------
// Problem constants (from the reference)
// ---------------------------------------------------------------------------
#define N_MOLS       64
#define N_CONFS_PER  10
#define A_ATOMS      40
#define N_CONFS      (N_MOLS * N_CONFS_PER)          // 640
#define N_ATOMS      (N_CONFS * A_ATOMS)             // 25600
#define N_EDGES      409600
#define DD           256
#define FF           256
#define NG           32
#define LL           4
#define MOL_BASIS    512
#define HH           ((DD + MOL_BASIS) / 2)          // 384
#define CUTOFF       5.0f
#define LOG2F_C      0.69314718055994530942f
#define EDGE_TILES   (N_EDGES / 16)                  // 25600

typedef __bf16 bhalf_t;
typedef __attribute__((ext_vector_type(16))) __bf16 v16bf;
typedef __attribute__((ext_vector_type(8)))  float  v8f;

// shifted softplus: log(1+exp(x)) - log(2), numerically stable
__device__ __forceinline__ float sspf(float x) {
    float ax = fabsf(x);
    return fmaxf(x, 0.0f) + __logf(1.0f + __expf(-ax)) - LOG2F_C;
}

// ---------------------------------------------------------------------------
// Weight repack: fp32 row-major W[K][N]  ->  bf16 in WMMA B-lane layout.
// Tile (kt,nt) covers K rows [kt*32,kt*32+32), cols [nt*16,nt*16+16).
// Within a tile: 32 lanes x 16 bf16 each. Lane holds column (lane&15),
// khalf = lane>>4; element pair 2v,2v+1 holds K = kt*32 + 2v + (v>=4?8:0)
// + 8*khalf (+0,+1)  — matches the 16-bit 16x32 VGPR striping in the ISA.
// ---------------------------------------------------------------------------
__global__ void pack_b_kernel(const float* __restrict__ W, bhalf_t* __restrict__ out,
                              int K, int N) {
    int idx = blockIdx.x * blockDim.x + threadIdx.x;
    int total = K * N;
    if (idx >= total) return;
    int e     = idx & 15;
    int lane  = (idx >> 4) & 31;
    int tile  = idx >> 9;
    int ntiles = N >> 4;
    int kt = tile / ntiles, nt = tile % ntiles;
    int v = e >> 1, p = e & 1;
    int khalf = lane >> 4;
    int k   = kt * 32 + 2 * v + (v >= 4 ? 8 : 0) + 8 * khalf + p;
    int col = nt * 16 + (lane & 15);
    out[idx] = (bhalf_t)W[(size_t)k * N + col];
}

// ---------------------------------------------------------------------------
// Precompute gaussian edge features e_g directly in WMMA A-lane layout.
// One thread per (edge-tile, lane): lane owns edge row (lane&15) of its tile
// and 16 K-values (khalf striping), so it needs only ONE distance.
// ---------------------------------------------------------------------------
__global__ void eg_pack_kernel(const float* __restrict__ xyz,
                               const int* __restrict__ nbr,
                               bhalf_t* __restrict__ out) {
    int idx = blockIdx.x * blockDim.x + threadIdx.x;   // tile*32 + lane
    if (idx >= EDGE_TILES * 32) return;
    int lane = idx & 31;
    int tile = idx >> 5;
    int edge = tile * 16 + (lane & 15);
    int i0 = nbr[2 * edge], i1 = nbr[2 * edge + 1];
    float dx = xyz[3 * i0 + 0] - xyz[3 * i1 + 0];
    float dy = xyz[3 * i0 + 1] - xyz[3 * i1 + 1];
    float dz = xyz[3 * i0 + 2] - xyz[3 * i1 + 2];
    float d  = sqrtf(dx * dx + dy * dy + dz * dz);
    const float width = CUTOFF / (float)(NG - 1);
    int khalf = lane >> 4;
    v16bf ov;
#pragma unroll
    for (int v = 0; v < 8; ++v) {
        int kb = 2 * v + (v >= 4 ? 8 : 0) + 8 * khalf;
#pragma unroll
        for (int p = 0; p < 2; ++p) {
            float off = (float)(kb + p) * width;
            float t = (d - off) / width;
            ov[2 * v + p] = (bhalf_t)__expf(-0.5f * t * t);
        }
    }
    *(v16bf*)(out + (size_t)idx * 16) = ov;
}

// r = emb[z]
__global__ void gather_emb_kernel(const int* __restrict__ z,
                                  const float* __restrict__ emb,
                                  float* __restrict__ r) {
    int idx = blockIdx.x * blockDim.x + threadIdx.x;
    if (idx >= N_ATOMS * DD) return;
    int a = idx >> 8, d = idx & 255;
    r[idx] = emb[(size_t)z[a] * DD + d];
}

// ---------------------------------------------------------------------------
// Generic WMMA GEMM:  C = act(A @ B + bias) [+ resid]
//   A: M x K fp32 row-major (converted to bf16 on load, A-lane layout)
//   B: packed bf16 (pack_b_kernel layout), bias: N fp32
//   Each wave computes a 16x64 output tile (4 n-tiles), K-loop in steps of 32.
// ---------------------------------------------------------------------------
__global__ void gemm_bf16_kernel(const float* __restrict__ Amat,
                                 const bhalf_t* __restrict__ Bp,
                                 const float* __restrict__ bias,
                                 const float* __restrict__ resid,
                                 float* __restrict__ C,
                                 int M, int N, int K, int act) {
    int wid  = (blockIdx.x * blockDim.x + threadIdx.x) >> 5;
    int lane = threadIdx.x & 31;
    int ngroups = N >> 6;            // groups of 4 n-tiles
    int mtiles  = M >> 4;
    if (wid >= mtiles * ngroups) return;
    int mt = wid / ngroups, ng = wid % ngroups;
    int ntiles = N >> 4;
    int ktiles = K >> 5;
    int row    = mt * 16 + (lane & 15);
    int khalf  = lane >> 4;

    const v8f zero = {0, 0, 0, 0, 0, 0, 0, 0};
    v8f acc[4];
#pragma unroll
    for (int t = 0; t < 4; ++t) acc[t] = zero;

    for (int kt = 0; kt < ktiles; ++kt) {
        v16bf a;
        const float* Ap = Amat + (size_t)row * K + kt * 32;
#pragma unroll
        for (int v = 0; v < 8; ++v) {
            int kb = 2 * v + (v >= 4 ? 8 : 0) + 8 * khalf;
            a[2 * v]     = (bhalf_t)Ap[kb];
            a[2 * v + 1] = (bhalf_t)Ap[kb + 1];
        }
        const bhalf_t* Bt = Bp + ((size_t)(kt * ntiles + ng * 4) * 32 + lane) * 16;
        v16bf b0 = *(const v16bf*)(Bt);
        v16bf b1 = *(const v16bf*)(Bt + 512);
        v16bf b2 = *(const v16bf*)(Bt + 1024);
        v16bf b3 = *(const v16bf*)(Bt + 1536);
        acc[0] = __builtin_amdgcn_wmma_f32_16x16x32_bf16(false, a, false, b0, (short)0, acc[0], false, false);
        acc[1] = __builtin_amdgcn_wmma_f32_16x16x32_bf16(false, a, false, b1, (short)0, acc[1], false, false);
        acc[2] = __builtin_amdgcn_wmma_f32_16x16x32_bf16(false, a, false, b2, (short)0, acc[2], false, false);
        acc[3] = __builtin_amdgcn_wmma_f32_16x16x32_bf16(false, a, false, b3, (short)0, acc[3], false, false);
    }

#pragma unroll
    for (int t = 0; t < 4; ++t) {
        int nt  = ng * 4 + t;
        int col = nt * 16 + (lane & 15);
        float bv = bias ? bias[col] : 0.0f;
#pragma unroll
        for (int g = 0; g < 8; ++g) {
            int m = mt * 16 + g + 8 * khalf;     // C/D layout: VGPR g -> M=g(+8)
            float val = acc[t][g] + bv;
            if (act == 1) val = sspf(val);
            if (resid) val += resid[(size_t)m * N + col];
            C[(size_t)m * N + col] = val;
        }
    }
}

// ---------------------------------------------------------------------------
// Fused edge kernel (per layer):
//   h  = ssp(e_g_tile @ We1 + be1)        (2 WMMAs, 16x32)
//   ef = h @ We2 + be2                    (16 WMMAs, done n-tile at a time)
//   agg[a0] += rn[a1]*ef ; agg[a1] += rn[a0]*ef   (fp32 global atomics)
// h is bounced through LDS to convert C/D layout -> A layout (wave-local).
// ---------------------------------------------------------------------------
__global__ void edge_kernel(const bhalf_t* __restrict__ eg_pack,
                            const bhalf_t* __restrict__ we1p,
                            const float* __restrict__ be1,
                            const bhalf_t* __restrict__ we2p,
                            const float* __restrict__ be2,
                            const int* __restrict__ nbr,
                            const float* __restrict__ rn,
                            float* __restrict__ agg) {
    __shared__ float lds[8 * 16 * 32];                 // 2KB per wave, 8 waves
    int wib  = threadIdx.x >> 5;
    int lane = threadIdx.x & 31;
    int tile = blockIdx.x * 8 + wib;
    if (tile >= EDGE_TILES) return;                    // uniform per wave
    int colx  = lane & 15;
    int khalf = lane >> 4;
    const v8f zero = {0, 0, 0, 0, 0, 0, 0, 0};

    // A = e_g tile (already in A-lane layout)
    v16bf a = *(const v16bf*)(eg_pack + ((size_t)tile * 32 + lane) * 16);
    v16bf w0 = *(const v16bf*)(we1p + (size_t)lane * 16);
    v16bf w1 = *(const v16bf*)(we1p + 512 + (size_t)lane * 16);
    v8f h0 = __builtin_amdgcn_wmma_f32_16x16x32_bf16(false, a, false, w0, (short)0, zero, false, false);
    v8f h1 = __builtin_amdgcn_wmma_f32_16x16x32_bf16(false, a, false, w1, (short)0, zero, false, false);

    float b0 = be1[colx];
    float b1 = be1[16 + colx];
    float* my = lds + wib * 512;
#pragma unroll
    for (int g = 0; g < 8; ++g) {
        int m = g + 8 * khalf;                         // C layout row
        my[m * 32 + colx]      = sspf(h0[g] + b0);
        my[m * 32 + 16 + colx] = sspf(h1[g] + b1);
    }
    asm volatile("s_wait_dscnt 0" ::: "memory");       // wave-local LDS RAW

    // reload h in A-lane layout (row = colx is this lane's edge row)
    v16bf ha;
#pragma unroll
    for (int v = 0; v < 8; ++v) {
        int kb = 2 * v + (v >= 4 ? 8 : 0) + 8 * khalf;
        ha[2 * v]     = (bhalf_t)my[colx * 32 + kb];
        ha[2 * v + 1] = (bhalf_t)my[colx * 32 + kb + 1];
    }

    // edge endpoints for this lane's 8 output rows (C layout rows)
    int e0[8], e1[8];
#pragma unroll
    for (int g = 0; g < 8; ++g) {
        int edge = tile * 16 + g + 8 * khalf;
        e0[g] = nbr[2 * edge];
        e1[g] = nbr[2 * edge + 1];
    }

#pragma unroll
    for (int nt = 0; nt < 16; ++nt) {
        v16bf bw = *(const v16bf*)(we2p + ((size_t)nt * 32 + lane) * 16);
        v8f ef = __builtin_amdgcn_wmma_f32_16x16x32_bf16(false, ha, false, bw, (short)0, zero, false, false);
        int c = nt * 16 + colx;
        float bb = be2[c];
#pragma unroll
        for (int g = 0; g < 8; ++g) {
            float vv = ef[g] + bb;
            float r1 = rn[(size_t)e1[g] * FF + c];
            float r0 = rn[(size_t)e0[g] * FF + c];
            unsafeAtomicAdd(agg + (size_t)e0[g] * FF + c, r1 * vv);
            unsafeAtomicAdd(agg + (size_t)e1[g] * FF + c, r0 * vv);
        }
    }
}

// conf_fp[c][d] = sum_a r[c*40+a][d]
__global__ void conf_reduce_kernel(const float* __restrict__ r, float* __restrict__ conf_fp) {
    int idx = blockIdx.x * blockDim.x + threadIdx.x;
    if (idx >= N_CONFS * DD) return;
    int c = idx >> 8, dch = idx & 255;
    float s = 0.0f;
#pragma unroll 8
    for (int a = 0; a < A_ATOMS; ++a)
        s += r[((size_t)c * A_ATOMS + a) * DD + dch];
    conf_fp[idx] = s;
}

// mol_fp[m][b] = sum_c molconf[m*10+c][b] * bw[m*10+c]
__global__ void mol_reduce_kernel(const float* __restrict__ molconf,
                                  const float* __restrict__ bw,
                                  float* __restrict__ molfp) {
    int idx = blockIdx.x * blockDim.x + threadIdx.x;
    if (idx >= N_MOLS * MOL_BASIS) return;
    int m = idx >> 9, b = idx & 511;
    float s = 0.0f;
#pragma unroll
    for (int c = 0; c < N_CONFS_PER; ++c) {
        int conf = m * N_CONFS_PER + c;
        s += molconf[(size_t)conf * MOL_BASIS + b] * bw[conf];
    }
    molfp[idx] = s;
}

// out[m] = sigmoid(hid[m] . Wr2 + br2)
__global__ void final_kernel(const float* __restrict__ hid,
                             const float* __restrict__ Wr2,
                             const float* __restrict__ br2,
                             float* __restrict__ out) {
    int m = blockIdx.x * blockDim.x + threadIdx.x;
    if (m >= N_MOLS) return;
    float s = br2[0];
    for (int k = 0; k < 256; ++k) s += hid[(size_t)m * 256 + k] * Wr2[k];
    out[m] = 1.0f / (1.0f + __expf(-s));
}

// ---------------------------------------------------------------------------
// Host launcher
// ---------------------------------------------------------------------------
static inline void launch_gemm(const float* A, const bhalf_t* Bp, const float* bias,
                               const float* resid, float* C,
                               int M, int N, int K, int act, hipStream_t s) {
    int waves  = (M / 16) * (N / 64);
    int blocks = (waves + 7) / 8;
    gemm_bf16_kernel<<<blocks, 256, 0, s>>>(A, Bp, bias, resid, C, M, N, K, act);
}

extern "C" void kernel_launch(void* const* d_in, const int* in_sizes, int n_in,
                              void* d_out, int out_size, void* d_ws, size_t ws_size,
                              hipStream_t stream) {
    const int*   z    = (const int*)d_in[0];
    const float* xyz  = (const float*)d_in[1];
    const int*   nbr  = (const int*)d_in[2];
    const float* bw   = (const float*)d_in[3];
    const float* emb  = (const float*)d_in[4];
    const float* We1  = (const float*)d_in[5];
    const float* be1  = (const float*)d_in[6];
    const float* We2  = (const float*)d_in[7];
    const float* be2  = (const float*)d_in[8];
    const float* Wn   = (const float*)d_in[9];
    const float* bn   = (const float*)d_in[10];
    const float* Wu1  = (const float*)d_in[11];
    const float* bu1  = (const float*)d_in[12];
    const float* Wu2  = (const float*)d_in[13];
    const float* bu2  = (const float*)d_in[14];
    const float* Wm1  = (const float*)d_in[15];
    const float* bm1  = (const float*)d_in[16];
    const float* Wm2  = (const float*)d_in[17];
    const float* bm2  = (const float*)d_in[18];
    const float* Wr1  = (const float*)d_in[19];
    const float* br1  = (const float*)d_in[20];
    const float* Wr2  = (const float*)d_in[21];
    const float* br2  = (const float*)d_in[22];
    float* out = (float*)d_out;

    // ---- workspace bump allocator ----
    char*  ws  = (char*)d_ws;
    size_t off = 0;
    auto alloc = [&](size_t bytes) -> char* {
        char* p = ws + off;
        off = (off + bytes + 255) & ~(size_t)255;
        return p;
    };
    bhalf_t* eg_p   = (bhalf_t*)alloc((size_t)EDGE_TILES * 512 * sizeof(bhalf_t));
    float*   r      = (float*)alloc((size_t)N_ATOMS * DD * sizeof(float));
    float*   rn     = (float*)alloc((size_t)N_ATOMS * DD * sizeof(float)); // reused as t
    float*   agg    = (float*)alloc((size_t)N_ATOMS * DD * sizeof(float));
    float*   conf   = (float*)alloc((size_t)N_CONFS * DD * sizeof(float));
    float*   t2     = (float*)alloc((size_t)N_CONFS * HH * sizeof(float));
    float*   mconf  = (float*)alloc((size_t)N_CONFS * MOL_BASIS * sizeof(float));
    float*   mfp    = (float*)alloc((size_t)N_MOLS * MOL_BASIS * sizeof(float));
    float*   hid    = (float*)alloc((size_t)N_MOLS * 256 * sizeof(float));
    bhalf_t* wn_p   = (bhalf_t*)alloc((size_t)LL * DD * FF * sizeof(bhalf_t));
    bhalf_t* wu1_p  = (bhalf_t*)alloc((size_t)LL * FF * DD * sizeof(bhalf_t));
    bhalf_t* wu2_p  = (bhalf_t*)alloc((size_t)LL * DD * DD * sizeof(bhalf_t));
    bhalf_t* we1_p  = (bhalf_t*)alloc((size_t)LL * NG * NG * sizeof(bhalf_t));
    bhalf_t* we2_p  = (bhalf_t*)alloc((size_t)LL * NG * FF * sizeof(bhalf_t));
    bhalf_t* wm1_p  = (bhalf_t*)alloc((size_t)DD * HH * sizeof(bhalf_t));
    bhalf_t* wm2_p  = (bhalf_t*)alloc((size_t)HH * MOL_BASIS * sizeof(bhalf_t));
    bhalf_t* wr1_p  = (bhalf_t*)alloc((size_t)MOL_BASIS * 256 * sizeof(bhalf_t));

    // ---- repack weights (deterministic, every call) ----
    auto pack = [&](const float* W, bhalf_t* O, int K, int N) {
        int total = K * N;
        pack_b_kernel<<<(total + 255) / 256, 256, 0, stream>>>(W, O, K, N);
    };
    for (int l = 0; l < LL; ++l) {
        pack(Wn  + (size_t)l * DD * FF, wn_p  + (size_t)l * DD * FF, DD, FF);
        pack(Wu1 + (size_t)l * FF * DD, wu1_p + (size_t)l * FF * DD, FF, DD);
        pack(Wu2 + (size_t)l * DD * DD, wu2_p + (size_t)l * DD * DD, DD, DD);
        pack(We1 + (size_t)l * NG * NG, we1_p + (size_t)l * NG * NG, NG, NG);
        pack(We2 + (size_t)l * NG * FF, we2_p + (size_t)l * NG * FF, NG, FF);
    }
    pack(Wm1, wm1_p, DD, HH);
    pack(Wm2, wm2_p, HH, MOL_BASIS);
    pack(Wr1, wr1_p, MOL_BASIS, 256);

    // ---- precompute edge gaussians (A-layout) and initial embeddings ----
    eg_pack_kernel<<<(EDGE_TILES * 32 + 255) / 256, 256, 0, stream>>>(xyz, nbr, eg_p);
    gather_emb_kernel<<<(N_ATOMS * DD + 255) / 256, 256, 0, stream>>>(z, emb, r);

    // ---- message-passing layers ----
    for (int l = 0; l < LL; ++l) {
        // rn = r @ Wn[l] + bn[l]
        launch_gemm(r, wn_p + (size_t)l * DD * FF, bn + (size_t)l * FF,
                    nullptr, rn, N_ATOMS, FF, DD, 0, stream);
        hipMemsetAsync(agg, 0, (size_t)N_ATOMS * DD * sizeof(float), stream);
        // fused edge MLP + scatter
        edge_kernel<<<EDGE_TILES / 8, 256, 0, stream>>>(
            eg_p,
            we1_p + (size_t)l * NG * NG, be1 + (size_t)l * NG,
            we2_p + (size_t)l * NG * FF, be2 + (size_t)l * FF,
            nbr, rn, agg);
        // t = ssp(agg @ Wu1[l] + bu1[l])   (reuse rn buffer)
        launch_gemm(agg, wu1_p + (size_t)l * FF * DD, bu1 + (size_t)l * DD,
                    nullptr, rn, N_ATOMS, DD, FF, 1, stream);
        // r = r + (t @ Wu2[l] + bu2[l])
        launch_gemm(rn, wu2_p + (size_t)l * DD * DD, bu2 + (size_t)l * DD,
                    r, r, N_ATOMS, DD, DD, 0, stream);
    }

    // ---- readout ----
    conf_reduce_kernel<<<(N_CONFS * DD + 255) / 256, 256, 0, stream>>>(r, conf);
    launch_gemm(conf, wm1_p, bm1, nullptr, t2, N_CONFS, HH, DD, 1, stream);
    launch_gemm(t2, wm2_p, bm2, nullptr, mconf, N_CONFS, MOL_BASIS, HH, 0, stream);
    mol_reduce_kernel<<<(N_MOLS * MOL_BASIS + 255) / 256, 256, 0, stream>>>(mconf, bw, mfp);
    launch_gemm(mfp, wr1_p, br1, nullptr, hid, N_MOLS, 256, MOL_BASIS, 1, stream);
    final_kernel<<<1, 64, 0, stream>>>(hid, Wr2, br2, out);
}
